// LSTMRegressor_67293547594335
// MI455X (gfx1250) — matrix operand order repaired and checked by reading, not compiled, not run on hardware
//
#include <hip/hip_runtime.h>

typedef __attribute__((ext_vector_type(16))) _Float16 v16h;
typedef __attribute__((ext_vector_type(8)))  _Float16 v8h;
typedef __attribute__((ext_vector_type(8)))  float    v8f;
typedef __attribute__((ext_vector_type(2)))  float    v2f;

constexpr int kH = 32;      // hidden size
constexpr int kT = 512;     // timesteps
constexpr int kB = 4096;    // batch
constexpr int kWavesPerWG = 4;
constexpr int kWave = 32;

// Native CDNA5 V_TANH_F32 if the toolchain exposes it; exp2-based fallback otherwise.
#if __has_builtin(__builtin_amdgcn_tanhf)
__device__ __forceinline__ float fast_tanh(float x) { return __builtin_amdgcn_tanhf(x); }
#else
__device__ __forceinline__ float fast_tanh(float x) {
    return 2.0f * __builtin_amdgcn_rcpf(1.0f + __builtin_amdgcn_exp2f(-2.8853900817779268f * x)) - 1.0f;
}
#endif
__device__ __forceinline__ float fast_sigmoid(float x) {
    // sigmoid(x) = 0.5*tanh(x/2) + 0.5  -> mul + tanh + fma (1 TRANS op)
    return __builtin_fmaf(0.5f, fast_tanh(0.5f * x), 0.5f);
}

__global__ __launch_bounds__(kWavesPerWG * kWave)
void lstm_wmma_kernel(const float* __restrict__ x,     // [B, T, 1]
                      const float* __restrict__ W_ih,  // [4H, 1]
                      const float* __restrict__ W_hh,  // [4H, H]
                      const float* __restrict__ b_ih,  // [4H]
                      const float* __restrict__ b_hh,  // [4H]
                      const float* __restrict__ W_fc,  // [1, H]
                      const float* __restrict__ b_fc,  // [1]
                      float* __restrict__ out)         // [B, 1]
{
    __shared__ __align__(16) _Float16 hbuf[kWavesPerWG][16 * kH]; // per-wave h relayout buffer
    __shared__ __align__(16) float    fbuf[kWavesPerWG][16 * kH]; // final h (f32) for output dot

    const int lane    = threadIdx.x & 31;
    const int wv      = threadIdx.x >> 5;
    const int b0      = (blockIdx.x * kWavesPerWG + wv) * 16; // 16 batch rows per wave
    const int lo16    = lane & 15;
    const bool lowHalf = lane < 16;

    // ---------------- constant WMMA fragments (resident for whole kernel) ----------
    // B fragments of W_hh^T (32x16 f16 per 16-col tile), contiguous-K B layout:
    //   lanes 0-15: col n = lane,  K = 0..15 (2 per VGPR); lanes 16-31: col n = lane-16, K = 16..31
    v16h Bh[8];
    // Rank-1 x-update B fragments (4x16 f32): lanes 0-15 VGPR0 = row K=0 = W_ih[g]; rest 0
    v2f  Bx[8];
    // Persistent bias C operands (b_ih + b_hh broadcast across the 8 D rows)
    v8f  CbiasV[8];
    {
        const int k0 = lowHalf ? 0 : 16;
        for (int j = 0; j < 8; ++j) {
            const int g = j * 16 + lo16;
            const float* wrow = W_hh + g * kH + k0;
            v16h bh;
            #pragma unroll
            for (int v = 0; v < 16; ++v) bh[v] = (_Float16)wrow[v];
            Bh[j] = bh;
            v2f bx;
            bx.x = lowHalf ? W_ih[g] : 0.0f;  // K=0 row (low half); K=2 row (high half) is zero
            bx.y = 0.0f;                       // K=1 / K=3 rows (zero)
            Bx[j] = bx;
            const float bb = b_ih[g] + b_hh[g];
            v8f cb;
            #pragma unroll
            for (int r = 0; r < 8; ++r) cb[r] = bb;
            CbiasV[j] = cb;
        }
    }

    // ---------------- recurrent state ----------------
    v16h Ah;                                  // A fragment of h (16x32 f16), zero at t=0
    #pragma unroll
    for (int v = 0; v < 16; ++v) Ah[v] = (_Float16)0.0f;

    // c and h in C/D layout: [k-tile][VGPR r]; lane half selects rows r / r+8, col = 16k + lo16
    float cst[2][8];
    float hst[2][8];
    #pragma unroll
    for (int k = 0; k < 2; ++k)
        #pragma unroll
        for (int r = 0; r < 8; ++r) { cst[k][r] = 0.0f; hst[k][r] = 0.0f; }

    _Float16* hb = hbuf[wv];
    const long xoff  = (long)(b0 + lo16) * kT;
    const int  mbase = lowHalf ? 0 : 8;
    const int  c0h   = lowHalf ? 0 : 16;

    float xl = x[xoff];                       // software-pipelined x load (t = 0)

    #pragma unroll 1
    for (int t = 0; t < kT; ++t) {
        v2f Ax;
        Ax.x = lowHalf ? xl : 0.0f;  // A 16x4 f32: lanes 0-15 VGPR0 = K=0 col = x[m]; rest 0
        Ax.y = 0.0f;

        // gates = h @ W_hh^T + bias + x (x) W_ih   -- all through WMMA
        v8f D[8];
        #pragma unroll
        for (int j = 0; j < 8; ++j) {
            v8f d = __builtin_amdgcn_wmma_f32_16x16x32_f16(
                        false, Ah, false, Bh[j], (short)0, CbiasV[j], false, false);
            d = __builtin_amdgcn_wmma_f32_16x16x4_f32(
                        false, Ax, false, Bx[j], (short)0, d, false, false);
            D[j] = d;
        }

        // Prefetch next timestep's x while the gate math below covers the latency
        {
            const int tn = (t + 1 < kT) ? (t + 1) : t;
            xl = x[xoff + tn];
        }

        // LSTM cell update, elementwise in C/D layout. PyTorch gate order: i, f, g, o
        #pragma unroll
        for (int k = 0; k < 2; ++k) {
            #pragma unroll
            for (int r = 0; r < 8; ++r) {
                const float iv = fast_sigmoid(D[0 + k][r]);
                const float fv = fast_sigmoid(D[2 + k][r]);
                const float gv = fast_tanh   (D[4 + k][r]);
                const float ov = fast_sigmoid(D[6 + k][r]);
                const float cv = fv * cst[k][r] + iv * gv;
                cst[k][r] = cv;
                hst[k][r] = ov * fast_tanh(cv);
            }
        }

        // Relayout h: D layout -> row-major f16 in LDS -> A layout (wave-private, in-order DS)
        #pragma unroll
        for (int k = 0; k < 2; ++k)
            #pragma unroll
            for (int r = 0; r < 8; ++r)
                hb[(mbase + r) * kH + k * 16 + lo16] = (_Float16)hst[k][r];

        // A 16-bit layout: lanes 0-15: K=0..15; lanes 16-31: K=16..31 (matches B fragment order)
        // Load the two 16B chunks straight into the A-fragment register pair.
        v8h* ahalf = (v8h*)&Ah;
        ahalf[0] = *(const v8h*)(hb + lo16 * kH + c0h);
        ahalf[1] = *(const v8h*)(hb + lo16 * kH + c0h + 8);
    }

    // ---------------- final projection: out[b] = h_T[b,:] . W_fc + b_fc ----------------
    float* fb = fbuf[wv];
    #pragma unroll
    for (int k = 0; k < 2; ++k)
        #pragma unroll
        for (int r = 0; r < 8; ++r)
            fb[(mbase + r) * kH + k * 16 + lo16] = hst[k][r];

    if (lowHalf) {
        float acc = b_fc[0];
        #pragma unroll
        for (int kk = 0; kk < kH; ++kk) acc += fb[lo16 * kH + kk] * W_fc[kk];
        out[b0 + lo16] = acc;
    }
}

extern "C" void kernel_launch(void* const* d_in, const int* in_sizes, int n_in,
                              void* d_out, int out_size, void* d_ws, size_t ws_size,
                              hipStream_t stream) {
    const float* x    = (const float*)d_in[0];
    const float* W_ih = (const float*)d_in[1];
    const float* W_hh = (const float*)d_in[2];
    const float* b_ih = (const float*)d_in[3];
    const float* b_hh = (const float*)d_in[4];
    const float* W_fc = (const float*)d_in[5];
    const float* b_fc = (const float*)d_in[6];

    dim3 block(kWavesPerWG * kWave);               // 128 threads = 4 waves
    dim3 grid(kB / (16 * kWavesPerWG));            // 4096 / 64 = 64 workgroups
    lstm_wmma_kernel<<<grid, block, 0, stream>>>(x, W_ih, W_hh, b_ih, b_hh, W_fc, b_fc,
                                                 (float*)d_out);
}
